// MatchingNet_65515431133926
// MI455X (gfx1250) — matrix operand order repaired and checked by reading, not compiled, not run on hardware
//
#include <hip/hip_runtime.h>
#include <hip/hip_bf16.h>
#include <math.h>

typedef __bf16 bf16;
typedef __attribute__((ext_vector_type(16))) __bf16 v16bf;
typedef __attribute__((ext_vector_type(8)))  float  v8f;

#define LP   4801
#define LC   4800
#define DC   256
#define DF   128
#define NM   2000
#define HF   240
#define WF_  320
#define WCC  80
#define NCHUNK 16
#define VCHUNK 301   // 16*301 = 4816 >= 4801

// GEMM tiling: workgroup 64x96, 4 waves, wave tile 32x48 (2x3 WMMA frags)
#define TM 64
#define TN 96

union FragBF { uint4 q[2]; v16bf v; };

// CDNA5 async global->LDS path (probe-confirmed on this toolchain); sync fallback kept.
#if defined(__has_builtin)
#if __has_builtin(__builtin_amdgcn_global_load_async_to_lds_b128) && \
    __has_builtin(__builtin_amdgcn_s_wait_asynccnt)
#define USE_ASYNC_LDS 1
#endif
#endif
#ifndef USE_ASYNC_LDS
#define USE_ASYNC_LDS 0
#endif

#if USE_ASYNC_LDS
typedef int v4i_gcc __attribute__((vector_size(16)));
typedef __attribute__((address_space(1))) v4i_gcc* async_gptr;
typedef __attribute__((address_space(3))) v4i_gcc* async_lptr;
#endif

// ---------- scale by 1/sqrt(256) and convert fp32 -> bf16 ----------
__global__ void k_convert_bf16(const float* __restrict__ in, bf16* __restrict__ out, int count) {
  int i = blockIdx.x * blockDim.x + threadIdx.x;
  int stride = gridDim.x * blockDim.x;
  for (; i < count; i += stride)
    out[i] = (bf16)(in[i] * 0.0625f);
}

// ---------- coarse score GEMM: C = A * B^T via v_wmma_f32_16x16x32_bf16 ----------
__global__ void __launch_bounds__(128) k_score_gemm(const bf16* __restrict__ A,
                                                    const bf16* __restrict__ Bm,
                                                    float* __restrict__ C) {
  // Unified tile: rows [0,TM) = A tile, rows [TM,TM+TN) = B tile (10 KB total)
  __shared__ bf16 Tile[TM + TN][32];
  const int bat = blockIdx.z;
  const int gm  = blockIdx.y * TM;
  const int gn  = blockIdx.x * TN;
  const int t    = threadIdx.x;
  const int lane = t & 31;
  const int wv   = t >> 5;
  const int wr = wv >> 1, wc = wv & 1;    // 2x2 waves; wave tile 32 x 48
  const int lh = lane >> 4, lr = lane & 15;

  v8f acc[2][3] = {};

  const bf16* Abase    = A  + ((size_t)bat * LC + gm) * DC;
  const bf16* Bbase    = Bm + ((size_t)bat * LC + gn) * DC;
  const bf16* BbaseAdj = Bbase - (size_t)TM * DC;   // so rows >= TM index B directly

  for (int kk = 0; kk < DC; kk += 32) {
    // stage (64+96)x32 bf16 tile: 640 b128 chunks over 128 threads, 5 full rounds,
    // branchless base-pointer select -> no exec divergence around async loads.
#pragma unroll
    for (int idx = t; idx < (TM + TN) * 4; idx += 128) {
      int row = idx >> 2, q = idx & 3;          // q: which 8-bf16 quarter of the 32-k row
      const bf16* gbase = (row < TM) ? Abase : BbaseAdj;
      const bf16* gsrc  = gbase + (size_t)row * DC + (kk + q * 8);
      bf16* ldst        = &Tile[row][q * 8];
#if USE_ASYNC_LDS
      __builtin_amdgcn_global_load_async_to_lds_b128(
          (async_gptr)gsrc, (async_lptr)ldst, 0, 0);
#else
      *(uint4*)ldst = *(const uint4*)gsrc;
#endif
    }
#if USE_ASYNC_LDS
    __builtin_amdgcn_s_wait_asynccnt(0);
#endif
    __syncthreads();

    FragBF fa[2], fb[3];
#pragma unroll
    for (int a = 0; a < 2; ++a) {   // A: lanes 0-15 K{0..7,16..23}, lanes 16-31 K{8..15,24..31}
      const bf16* ap = &Tile[wr * 32 + a * 16 + lr][0];
      fa[a].q[0] = *(const uint4*)(ap + lh * 8);
      fa[a].q[1] = *(const uint4*)(ap + lh * 8 + 16);
    }
#pragma unroll
    for (int b2 = 0; b2 < 3; ++b2) { // B: lane = N, lanes 0-15 K{0..7,16..23}, lanes 16-31 K{8..15,24..31}
      const bf16* bp = &Tile[TM + wc * 48 + b2 * 16 + lr][0];
      fb[b2].q[0] = *(const uint4*)(bp + lh * 8);
      fb[b2].q[1] = *(const uint4*)(bp + lh * 8 + 16);
    }
#pragma unroll
    for (int a = 0; a < 2; ++a)
#pragma unroll
      for (int b2 = 0; b2 < 3; ++b2)
        acc[a][b2] = __builtin_amdgcn_wmma_f32_16x16x32_bf16(
            false, fa[a].v, false, fb[b2].v, (short)0, acc[a][b2], false, false);
    __syncthreads();
  }

  // D layout: VGPR r -> M = r + (lane>=16)*8, N = lane&15. Row stride 4801 (bins col).
  float* Cb = C + (size_t)bat * LP * LP;
#pragma unroll
  for (int a = 0; a < 2; ++a)
#pragma unroll
    for (int b2 = 0; b2 < 3; ++b2) {
      int colg  = gn + wc * 48 + b2 * 16 + lr;
      int rbase = gm + wr * 32 + a * 16 + lh * 8;
#pragma unroll
      for (int r = 0; r < 8; ++r)
        Cb[(size_t)(rbase + r) * LP + colg] = acc[a][b2][r];
    }
}

// ---------- fill dustbin row/col with alpha ----------
__global__ void k_fill_bins(float* __restrict__ C, const float* __restrict__ alpha_p) {
  int i = blockIdx.x * blockDim.x + threadIdx.x;
  if (i >= 2 * LP) return;
  float alpha = alpha_p[0];
  int b = i / LP, k = i % LP;
  float* Cb = C + (size_t)b * LP * LP;
  Cb[(size_t)k * LP + (LP - 1)] = alpha;
  Cb[(size_t)(LP - 1) * LP + k] = alpha;
}

__global__ void k_zero2(float* u, float* v, int n) {
  int i = blockIdx.x * blockDim.x + threadIdx.x;
  if (i < n) { u[i] = 0.f; v[i] = 0.f; }
}

// ---------- Sinkhorn u-update: u[b,i] = log_mu(i) - LSE_j(C[b,i,j] + v[b,j]) ----------
__global__ void k_sink_u(const float* __restrict__ C, const float* __restrict__ v,
                         float* __restrict__ u) {
  __shared__ float sm[256], ss[256];
  int row = blockIdx.x, b = blockIdx.y, t = threadIdx.x;
  const float* Crow = C + ((size_t)b * LP + row) * LP;
  const float* vb   = v + b * LP;
  float m = -INFINITY, s = 0.f;
  for (int j = t; j < LP; j += 256) {
    float x = Crow[j] + vb[j];
    if (x > m) { s = s * __expf(m - x) + 1.f; m = x; }
    else        s += __expf(x - m);
  }
  sm[t] = m; ss[t] = s;
  __syncthreads();
  for (int o = 128; o > 0; o >>= 1) {
    if (t < o) {
      float m2 = sm[t + o], s2 = ss[t + o];
      float M = fmaxf(sm[t], m2);
      ss[t] = ss[t] * __expf(sm[t] - M) + s2 * __expf(m2 - M);
      sm[t] = M;
    }
    __syncthreads();
  }
  if (t == 0) {
    const float norm = -logf(9600.f);
    float logmu = (row < LC) ? norm : (logf((float)LC) + norm);
    u[b * LP + row] = logmu - (sm[0] + logf(ss[0]));
  }
}

// ---------- Sinkhorn v-update stage 1: per (colblock, rowchunk) online LSE partials ----------
__global__ void k_sink_vpart(const float* __restrict__ C, const float* __restrict__ u,
                             float2* __restrict__ part) {
  int j = blockIdx.x * 256 + threadIdx.x;
  int chunk = blockIdx.y;
  int b = blockIdx.z;
  if (j >= LP) return;
  int i0 = chunk * VCHUNK;
  int i1 = i0 + VCHUNK; if (i1 > LP) i1 = LP;
  const float* Cb = C + (size_t)b * LP * LP;
  const float* ub = u + b * LP;
  float m = -INFINITY, s = 0.f;
  for (int i = i0; i < i1; ++i) {
    float x = Cb[(size_t)i * LP + j] + ub[i];
    if (x > m) { s = s * __expf(m - x) + 1.f; m = x; }
    else        s += __expf(x - m);
  }
  part[((size_t)(b * NCHUNK + chunk)) * LP + j] = make_float2(m, s);
}

// ---------- Sinkhorn v-update stage 2: combine partials ----------
__global__ void k_sink_vcombine(const float2* __restrict__ part, float* __restrict__ v) {
  int idx = blockIdx.x * 256 + threadIdx.x;
  if (idx >= 2 * LP) return;
  int b = idx / LP, j = idx % LP;
  float M = -INFINITY, S = 0.f;
#pragma unroll
  for (int c = 0; c < NCHUNK; ++c) {
    float2 p = part[((size_t)(b * NCHUNK + c)) * LP + j];
    float nm = fmaxf(M, p.x);
    S = S * __expf(M - nm) + p.y * __expf(p.x - nm);
    M = nm;
  }
  const float norm = -logf(9600.f);
  float lognu = (j < LC) ? norm : (logf((float)LC) + norm);
  v[idx] = lognu - (M + logf(S));
}

// ---------- Z - norm, in place over couplings ----------
__global__ void k_finalize(float* __restrict__ C, const float* __restrict__ u,
                           const float* __restrict__ v) {
  size_t idx = (size_t)blockIdx.x * blockDim.x + threadIdx.x;
  const size_t per = (size_t)LP * LP;
  if (idx >= 2 * per) return;
  int b = (int)(idx / per);
  size_t rem = idx - (size_t)b * per;
  int i = (int)(rem / LP), j = (int)(rem % LP);
  C[idx] += u[b * LP + i] + v[b * LP + j] + logf(9600.f);
}

// ---------- center projection: cproj[n] = (mdesc/16) @ W_proj + b_proj ----------
__global__ void __launch_bounds__(128) k_center_proj(
    const float* __restrict__ mdesc0, const float* __restrict__ mdesc1,
    const int* __restrict__ matches, const float* __restrict__ Wp,
    const float* __restrict__ bp, float* __restrict__ cproj) {
  __shared__ float rowLds[DC];
  int n = blockIdx.x, t = threadIdx.x;
  int mrow = (n < NM) ? n : (n - NM);
  int b   = matches[mrow * 3 + 0];
  int idx = (n < NM) ? matches[mrow * 3 + 1] : matches[mrow * 3 + 2];
  const float* src = (n < NM) ? mdesc0 : mdesc1;
  const float* r = src + ((size_t)b * LC + idx) * DC;
  rowLds[t]       = r[t]       * 0.0625f;
  rowLds[t + 128] = r[t + 128] * 0.0625f;
  __syncthreads();
  float acc = bp[t];
#pragma unroll 4
  for (int d = 0; d < DC; ++d) acc += rowLds[d] * Wp[d * DF + t];
  cproj[(size_t)n * DF + t] = acc;
}

// ---------- fine matching: gather windows, merge MLP, correlation softmax, moments ----------
__global__ void __launch_bounds__(128) k_fine_match(
    const float* __restrict__ fm0, const float* __restrict__ fm1,
    const int* __restrict__ matches, const float* __restrict__ Wm,
    const float* __restrict__ bm, const float* __restrict__ pos1d,
    const float* __restrict__ cproj,
    float* __restrict__ out_std, float* __restrict__ out_mk0, float* __restrict__ out_mk1) {
  __shared__ float fineLds[DF];
  __shared__ float cp0[DF], cp1[DF];
  __shared__ float d0c[DF];
  __shared__ float d1[25][DF];
  __shared__ float simLds[25];

  int n = blockIdx.x, t = threadIdx.x;
  int b  = matches[n * 3 + 0];
  int i0 = matches[n * 3 + 1];
  int i1 = matches[n * 3 + 2];
  int r0 = i0 / WCC, c0 = i0 % WCC;
  int r1 = i1 / WCC, c1 = i1 % WCC;

  cp0[t] = cproj[(size_t)n * DF + t];
  cp1[t] = cproj[(size_t)(n + NM) * DF + t];

  // d0 center window (w=12 -> wy=wx=2 -> y=r0*4, x=c0*4, always in bounds)
  fineLds[t] = fm0[((size_t)b * DF + t) * (HF * WF_) + (r0 * 4) * WF_ + (c0 * 4)];
  __syncthreads();
  {
    float acc = bm[t] + pos1d[12 * DF + t];
    for (int c = 0; c < DF; ++c) acc += fineLds[c] * Wm[c * DF + t];
    for (int c = 0; c < DF; ++c) acc += cp0[c] * Wm[(DF + c) * DF + t];
    d0c[t] = acc;
  }
  for (int w = 0; w < 25; ++w) {
    int wy = w / 5, wx = w % 5;
    int y = r1 * 4 + wy - 2;
    int x = c1 * 4 + wx - 2;
    __syncthreads();
    float val = 0.f;
    if (y >= 0 && y < HF && x >= 0 && x < WF_)
      val = fm1[((size_t)b * DF + t) * (HF * WF_) + y * WF_ + x];
    fineLds[t] = val;
    __syncthreads();
    float acc = bm[t] + pos1d[w * DF + t];
    for (int c = 0; c < DF; ++c) acc += fineLds[c] * Wm[c * DF + t];
    for (int c = 0; c < DF; ++c) acc += cp1[c] * Wm[(DF + c) * DF + t];
    d1[w][t] = acc;
  }
  __syncthreads();
  if (t < 25) {
    float s = 0.f;
    for (int c = 0; c < DF; ++c) s += d0c[c] * d1[t][c];
    simLds[t] = s * 0.08838834764831843f;   // 1/sqrt(128)
  }
  __syncthreads();
  if (t == 0) {
    float mx = -INFINITY;
    for (int m = 0; m < 25; ++m) mx = fmaxf(mx, simLds[m]);
    float heat[25]; float sum = 0.f;
    for (int m = 0; m < 25; ++m) { heat[m] = __expf(simLds[m] - mx); sum += heat[m]; }
    float inv = 1.f / sum;
    float cx = 0.f, cy = 0.f, vx = 0.f, vy = 0.f;
    for (int m = 0; m < 25; ++m) {
      float h  = heat[m] * inv;
      float gx = -1.f + 0.5f * (float)(m % 5);
      float gy = -1.f + 0.5f * (float)(m / 5);
      cx += h * gx; cy += h * gy;
      vx += h * gx * gx; vy += h * gy * gy;
    }
    vx -= cx * cx; vy -= cy * cy;
    float sd = sqrtf(fmaxf(vx, 1e-10f)) + sqrtf(fmaxf(vy, 1e-10f));
    out_std[n] = sd;
    out_mk0[n * 2 + 0] = (float)c0 * 8.f;
    out_mk0[n * 2 + 1] = (float)r0 * 8.f;
    out_mk1[n * 2 + 0] = (float)c1 * 8.f + cx * 4.f;
    out_mk1[n * 2 + 1] = (float)r1 * 8.f + cy * 4.f;
  }
}

extern "C" void kernel_launch(void* const* d_in, const int* in_sizes, int n_in,
                              void* d_out, int out_size, void* d_ws, size_t ws_size,
                              hipStream_t stream) {
  const float* mdesc0 = (const float*)d_in[0];
  const float* mdesc1 = (const float*)d_in[1];
  const float* fm0    = (const float*)d_in[2];
  const float* fm1    = (const float*)d_in[3];
  const int*   matches= (const int*)d_in[4];
  const float* alpha  = (const float*)d_in[5];
  const float* Wp     = (const float*)d_in[6];
  const float* bp     = (const float*)d_in[7];
  const float* Wm     = (const float*)d_in[8];
  const float* bm     = (const float*)d_in[9];
  const float* pos1d  = (const float*)d_in[10];

  float* scores  = (float*)d_out;                         // (2, 4801, 4801)
  float* out_std = scores + (size_t)2 * LP * LP;          // (2000,)
  float* out_mk0 = out_std + NM;                          // (2000, 2)
  float* out_mk1 = out_mk0 + 2 * NM;                      // (2000, 2)

  char* ws = (char*)d_ws;
  size_t off = 0;
  bf16* m0b = (bf16*)(ws + off);  off += (size_t)2 * LC * DC * sizeof(bf16);
  bf16* m1b = (bf16*)(ws + off);  off += (size_t)2 * LC * DC * sizeof(bf16);
  float* u  = (float*)(ws + off); off += (size_t)2 * LP * sizeof(float);
  float* v  = (float*)(ws + off); off += (size_t)2 * LP * sizeof(float);
  off = (off + 15) & ~(size_t)15;
  float2* vpart = (float2*)(ws + off); off += (size_t)2 * NCHUNK * LP * sizeof(float2);
  float* cproj  = (float*)(ws + off);  off += (size_t)2 * NM * DF * sizeof(float);

  const int cnt = 2 * LC * DC;
  k_convert_bf16<<<(cnt + 255) / 256, 256, 0, stream>>>(mdesc0, m0b, cnt);
  k_convert_bf16<<<(cnt + 255) / 256, 256, 0, stream>>>(mdesc1, m1b, cnt);

  dim3 gg(LC / TN, LC / TM, 2);                 // 50 x 75 x 2 tiles
  k_score_gemm<<<gg, 128, 0, stream>>>(m0b, m1b, scores);

  k_fill_bins<<<(2 * LP + 255) / 256, 256, 0, stream>>>(scores, alpha);
  k_zero2<<<(2 * LP + 255) / 256, 256, 0, stream>>>(u, v, 2 * LP);

  for (int it = 0; it < 50; ++it) {
    k_sink_u<<<dim3(LP, 2), 256, 0, stream>>>(scores, v, u);
    k_sink_vpart<<<dim3((LP + 255) / 256, NCHUNK, 2), 256, 0, stream>>>(scores, u, vpart);
    k_sink_vcombine<<<(2 * LP + 255) / 256, 256, 0, stream>>>(vpart, v);
  }

  size_t total = (size_t)2 * LP * LP;
  k_finalize<<<(unsigned)((total + 255) / 256), 256, 0, stream>>>(scores, u, v);

  k_center_proj<<<2 * NM, 128, 0, stream>>>(mdesc0, mdesc1, matches, Wp, bp, cproj);
  k_fine_match<<<NM, 128, 0, stream>>>(fm0, fm1, matches, Wm, bm, pos1d, cproj,
                                       out_std, out_mk0, out_mk1);
}
